// LocalMSA0_7103875907870
// MI455X (gfx1250) — matrix operand order repaired and checked by reading, not compiled, not run on hardware
//
#include <hip/hip_runtime.h>

typedef __attribute__((ext_vector_type(16))) _Float16 v16h;
typedef __attribute__((ext_vector_type(8)))  float    v8f;

#define DIM   192
#define HEADS 6
#define HD    32
#define NPIX  65536      // 256*256
#define BATCH 4
#define KDIM  192        // inner dim of both GEMMs (compile-time for full unroll)

// ---------------------------------------------------------------------------
// 1x1-conv GEMM:  Y[b,o,p] = sum_c W[o,c] * X[b,c,p]  (+ bias[o])
// Each workgroup owns a 64-pixel strip and the FULL K=192: X is staged in LDS
// once (24 KB f16) and reused across all M/64 output tiles, so global x/att
// traffic is exactly 1x. Weight tiles (64x192 f16, 24 KB) are restaged per
// M-tile (weights are L2-resident: w_qkv is only 0.44 MB).
// 128 threads = 4 waves; wave w owns output rows [16w, 16w+16) of each tile.
// K fully unrolled: 6 k-steps x 4 n-tiles = 24 WMMAs per M-tile.
// ---------------------------------------------------------------------------
__global__ __launch_bounds__(128) void gemm_wmma_kernel(
    const float* __restrict__ W,     // [M, KDIM] row-major
    const float* __restrict__ X,     // [B, KDIM, NPIX]
    float* __restrict__ Y,           // [B, M, NPIX]
    const float* __restrict__ bias,  // [M] or nullptr
    int M)
{
    __shared__ _Float16 sX[64 * KDIM];   // [p][c]  (B operand: [n][k])
    __shared__ _Float16 sA[64 * KDIM];   // [o][c]  (A operand: [m][k])

    const int tid    = threadIdx.x;
    const int waveId = tid >> 5;
    const int lane   = tid & 31;
    const int half   = lane >> 4;
    const int ln     = lane & 15;
    const int p0     = blockIdx.x * 64;
    const int b      = blockIdx.z;

    // stage the full 64 x 192 activation strip once (coalesced over p)
    #pragma unroll
    for (int q = 0; q < (64 * KDIM) / 128; ++q) {
        int idx = q * 128 + tid;
        int pp = idx & 63, c = idx >> 6;
        sX[pp * KDIM + c] =
            (_Float16)X[((size_t)b * KDIM + c) * (size_t)NPIX + (p0 + pp)];
    }

    const int mTiles = M / 64;
    for (int ot = 0; ot < mTiles; ++ot) {
        __syncthreads();   // sX ready (1st iter) / prior sA reads done
        const int o0 = ot * 64;
        // stage 64 x 192 weight tile (each thread: 96 contiguous floats)
        #pragma unroll
        for (int q = 0; q < (64 * KDIM) / 128; ++q) {
            int idx = tid * ((64 * KDIM) / 128) + q;
            sA[idx] = (_Float16)W[(size_t)o0 * KDIM + idx];
        }
        __syncthreads();

        v8f acc[4] = {};
        #pragma unroll
        for (int k0 = 0; k0 < KDIM; k0 += 32) {
            v16h a;
            #pragma unroll
            for (int e = 0; e < 16; ++e) {
                int kk = half * 16 + e;
                a[e] = sA[(waveId * 16 + ln) * KDIM + k0 + kk];
            }
            #pragma unroll
            for (int nt = 0; nt < 4; ++nt) {
                v16h bb;
                #pragma unroll
                for (int e = 0; e < 16; ++e)
                    bb[e] = sX[(nt * 16 + ln) * KDIM + k0 + half * 16 + e];
                acc[nt] = __builtin_amdgcn_wmma_f32_16x16x32_f16(
                    false, a, false, bb, (short)0, acc[nt], false, false);
            }
        }

        // store D: VGPR r -> m = r + 8*half, n = ln
        #pragma unroll
        for (int nt = 0; nt < 4; ++nt) {
            int p = p0 + nt * 16 + ln;
            #pragma unroll
            for (int r = 0; r < 8; ++r) {
                int o = o0 + waveId * 16 + r + 8 * half;
                float v = acc[nt][r];
                if (bias) v += bias[o];
                Y[((size_t)b * M + o) * (size_t)NPIX + p] = v;
            }
        }
    }
}

// ---------------------------------------------------------------------------
// Fused depthwise-3x3-conv + 8x8-window attention.
// One workgroup per (window, head). 128 threads = 4 waves.
//   1. load 96-channel x 10x10 halo patch of qkv into LDS
//   2. apply 3x3 depthwise conv -> Q[i][d], K[j][d], Vt[d][j] (f16)
//   3. sim = Q K^T via WMMA (4x4 tiles, K-dim = 32)
//   4. softmax(sim*scale + pos_emb) in fp32 LDS (patch buffer reused)
//   5. out = attn V via WMMA (2 K-steps of 32), store merged (b,c,h,w)
// ---------------------------------------------------------------------------
__global__ __launch_bounds__(128) void attn_wmma_kernel(
    const float* __restrict__ qkv,      // [B, 3*DIM, 256, 256]
    const float* __restrict__ w_dw,     // [3*DIM, 9]
    const float* __restrict__ pos_emb,  // [HEADS, 64, 64]
    float* __restrict__ yout)           // [B, DIM, 256, 256]
{
    __shared__ float    sPatch[96 * 100];  // halo patch; reused for sim/attn
    __shared__ float    sW[96 * 9];
    __shared__ _Float16 sQ[64 * 32];       // [i][d]
    __shared__ _Float16 sK[64 * 32];       // [j][d]
    __shared__ _Float16 sVt[32 * 64];      // [d][j]

    float*     sSim  = sPatch;                       // 64*64 fp32 (16 KB)
    _Float16*  sAttn = (_Float16*)(sPatch + 64*64);  // 64*64 f16  (8 KB, disjoint)

    const int tid  = threadIdx.x;
    const int wid  = blockIdx.x;
    const int wy   = wid >> 5, wx = wid & 31;
    const int head = blockIdx.y;
    const int b    = blockIdx.z;

    // 1. halo patch load (zero-padded)
    for (int idx = tid; idx < 96 * 100; idx += 128) {
        int ci = idx / 100, pix = idx % 100;
        int py = pix / 10, px = pix % 10;
        int m = ci >> 5, d = ci & 31;
        int ch = m * DIM + head * HD + d;
        int gy = wy * 8 - 1 + py, gx = wx * 8 - 1 + px;
        float v = 0.f;
        if ((unsigned)gy < 256u && (unsigned)gx < 256u)
            v = qkv[((size_t)b * (3 * DIM) + ch) * (size_t)NPIX + gy * 256 + gx];
        sPatch[idx] = v;
    }
    for (int idx = tid; idx < 96 * 9; idx += 128) {
        int ci = idx / 9, t = idx % 9;
        int m = ci >> 5, d = ci & 31;
        sW[idx] = w_dw[(m * DIM + head * HD + d) * 9 + t];
    }
    __syncthreads();

    // 2. depthwise conv -> Q/K/Vt fragments in f16
    for (int idx = tid; idx < 3 * 32 * 64; idx += 128) {
        int m   = idx / (32 * 64);
        int rem = idx % (32 * 64);
        int d = rem >> 6, i = rem & 63;
        int ry = i >> 3, rx = i & 7;
        int ci = m * 32 + d;
        float acc = 0.f;
        #pragma unroll
        for (int ky = 0; ky < 3; ++ky)
            #pragma unroll
            for (int kx = 0; kx < 3; ++kx)
                acc += sW[ci * 9 + ky * 3 + kx] *
                       sPatch[ci * 100 + (ry + ky) * 10 + (rx + kx)];
        _Float16 h = (_Float16)acc;
        if      (m == 0) sQ[i * 32 + d]  = h;
        else if (m == 1) sK[i * 32 + d]  = h;
        else             sVt[d * 64 + i] = h;
    }
    __syncthreads();   // patch reads done; sSim may now overwrite it

    const int waveId = tid >> 5;
    const int lane   = tid & 31;
    const int half   = lane >> 4;
    const int ln     = lane & 15;
    const int i0     = waveId * 16;

    // 3. sim = Q K^T  (A = Q[i][d], B = K as [n=j][k=d], one WMMA per tile)
    {
        v16h a;
        #pragma unroll
        for (int e = 0; e < 16; ++e) {
            int kk = half * 16 + e;
            a[e] = sQ[(i0 + ln) * 32 + kk];
        }
        #pragma unroll
        for (int jt = 0; jt < 4; ++jt) {
            v16h bb;
            #pragma unroll
            for (int e = 0; e < 16; ++e)
                bb[e] = sK[(jt * 16 + ln) * 32 + half * 16 + e];
            v8f c = {};
            c = __builtin_amdgcn_wmma_f32_16x16x32_f16(
                false, a, false, bb, (short)0, c, false, false);
            #pragma unroll
            for (int r = 0; r < 8; ++r)
                sSim[(i0 + r + 8 * half) * 64 + jt * 16 + ln] = c[r];
        }
    }
    __syncthreads();

    // 4. softmax over j, with scale and positional bias
    if (tid < 64) {
        const int row = tid;
        const float scale = 0.17677669529663687f;   // 32^-0.5
        const float* pe = pos_emb + ((size_t)head * 64 + row) * 64;
        float mx = -1e30f;
        for (int j = 0; j < 64; ++j) {
            float s = sSim[row * 64 + j] * scale + pe[j];
            sSim[row * 64 + j] = s;
            mx = fmaxf(mx, s);
        }
        float sum = 0.f;
        for (int j = 0; j < 64; ++j) {
            float e = __expf(sSim[row * 64 + j] - mx);
            sSim[row * 64 + j] = e;
            sum += e;
        }
        float inv = 1.f / sum;
        for (int j = 0; j < 64; ++j)
            sAttn[row * 64 + j] = (_Float16)(sSim[row * 64 + j] * inv);
    }
    __syncthreads();

    // 5. out = attn V  (A = attn[i][j], B = Vt[n=d][k=j], 2 K-steps of 32)
    #pragma unroll
    for (int dt = 0; dt < 2; ++dt) {
        v8f c = {};
        #pragma unroll
        for (int ks = 0; ks < 2; ++ks) {
            v16h a, bb;
            #pragma unroll
            for (int e = 0; e < 16; ++e) {
                int kk = half * 16 + e;
                a[e] = sAttn[(i0 + ln) * 64 + ks * 32 + kk];
            }
            #pragma unroll
            for (int e = 0; e < 16; ++e)
                bb[e] = sVt[(dt * 16 + ln) * 64 + ks * 32 + half * 16 + e];
            c = __builtin_amdgcn_wmma_f32_16x16x32_f16(
                false, a, false, bb, (short)0, c, false, false);
        }
        // store merged window layout: c = head*32 + d, (h,w) from window pos
        #pragma unroll
        for (int r = 0; r < 8; ++r) {
            int i   = i0 + r + 8 * half;
            int d   = dt * 16 + ln;
            int och = head * HD + d;
            int gy  = wy * 8 + (i >> 3), gx = wx * 8 + (i & 7);
            yout[((size_t)b * DIM + och) * (size_t)NPIX + gy * 256 + gx] = c[r];
        }
    }
}

// ---------------------------------------------------------------------------
extern "C" void kernel_launch(void* const* d_in, const int* in_sizes, int n_in,
                              void* d_out, int out_size, void* d_ws, size_t ws_size,
                              hipStream_t stream) {
    const float* x      = (const float*)d_in[0];
    const float* w_qkv  = (const float*)d_in[1];
    const float* w_dw   = (const float*)d_in[2];
    const float* pos    = (const float*)d_in[3];
    const float* w_proj = (const float*)d_in[4];
    const float* b_proj = (const float*)d_in[5];
    float* out = (float*)d_out;

    // workspace: qkv [B,576,NPIX] then attention output [B,192,NPIX]
    float* qkv = (float*)d_ws;
    float* att = qkv + (size_t)BATCH * 3 * DIM * NPIX;

    dim3 blk(128);
    gemm_wmma_kernel<<<dim3(NPIX / 64, 1, BATCH), blk, 0, stream>>>(
        w_qkv, x, qkv, nullptr, 3 * DIM);
    attn_wmma_kernel<<<dim3(1024, HEADS, BATCH), blk, 0, stream>>>(
        qkv, w_dw, pos, att);
    gemm_wmma_kernel<<<dim3(NPIX / 64, 1, BATCH), blk, 0, stream>>>(
        w_proj, att, out, b_proj, DIM);
}
